// BigVGAN24k_22771916603897
// MI455X (gfx1250) — matrix operand order repaired and checked by reading, not compiled, not run on hardware
//
#include <hip/hip_runtime.h>
#include <math.h>

// ---------------------------------------------------------------------------
// BigVGAN forward for MI455X (gfx1250), wave32 + WMMA bf16 16x16x32.
// ---------------------------------------------------------------------------

typedef __bf16 bf16_t;
typedef __attribute__((ext_vector_type(8)))  __bf16 v8bf;
typedef __attribute__((ext_vector_type(16))) __bf16 v16bf;
typedef __attribute__((ext_vector_type(8)))  float  v8f;

struct Filt { float f[12]; };

// ---------------- weight repack: f32 (Co,Ci,k) -> bf16 [j][co][ci(pad KP)] --
__global__ void repack_conv_k(const float* __restrict__ src, bf16_t* __restrict__ dst,
                              int Co, int Ci, int KP, int k, int n) {
  int idx = blockIdx.x * blockDim.x + threadIdx.x;
  if (idx >= n) return;
  int j  = idx / (Co * KP);
  int r  = idx - j * (Co * KP);
  int co = r / KP;
  int ci = r - co * KP;
  float v = (ci < Ci) ? src[((size_t)co * Ci + ci) * k + j] : 0.0f;
  dst[idx] = (bf16_t)v;
}

// transposed-conv weights f32 (Ci,Co,k) -> bf16 [phase r][tap p][co][ci(pad)]
// W[r][p][co][ci] = wt[ci][co][k-1-(j0(r)+u*p)],  j0(r) = (pad - r) mod u
__global__ void repack_tconv_k(const float* __restrict__ wt, bf16_t* __restrict__ dst,
                               int Ci, int Co, int KP, int k, int u, int P, int pad, int n) {
  int idx = blockIdx.x * blockDim.x + threadIdx.x;
  if (idx >= n) return;
  int per = P * Co * KP;
  int r   = idx / per;
  int r1  = idx - r * per;
  int p   = r1 / (Co * KP);
  int r2  = r1 - p * (Co * KP);
  int co  = r2 / KP;
  int ci  = r2 - co * KP;
  int j0  = ((pad - r) % u + u) % u;
  int j   = j0 + u * p;
  float v = (ci < Ci) ? wt[((size_t)ci * Co + co) * k + (k - 1 - j)] : 0.0f;
  dst[idx] = (bf16_t)v;
}

// ---------------- core: dilated 1-D conv as WMMA GEMM ----------------------
// out[b, co, tg*ostride+ooff] = bias[co] + res[..] +
//   sum_j sum_ci W[j][co][ci] * x[b, ci, tg + j*dil - padl]
// WG: 256 thr (8 waves). Tile: 32 co x 128 t. Each wave: 16 co x 32 t (2 acc).
// KP (padded Cin, multiple of 32) is a compile-time constant so the K-chunk
// loop fully unrolls into back-to-back WMMAs with constant LDS offsets.
template <int KP>
__global__ __launch_bounds__(256) void conv_wmma_k(
    const float* __restrict__ x, const bf16_t* __restrict__ w,
    const float* __restrict__ bias, const float* __restrict__ res,
    float* __restrict__ out,
    int Cin, int Lin, int Cout, int K, int dil, int padl,
    int Lq, int ostride, int ooff, int LoutFull)
{
  extern __shared__ char smem_raw[];
  bf16_t* sx = (bf16_t*)smem_raw;   // [window][KP] bf16, zero padded

  const int tid    = threadIdx.x;
  const int b      = blockIdx.z;
  const int t0     = blockIdx.x * 128;
  const int window = 128 + (K - 1) * dil;   // always >= 129

  // Stage x tile -> LDS (f32 -> bf16). Consecutive tid -> consecutive global t
  // (coalesced). Incremental ci/tt bookkeeping: one division at loop entry.
  {
    const int count = window * KP;
    const float* xb = x + (size_t)b * Cin * Lin;
    int ci = tid / window;
    int tt = tid - ci * window;
    for (int idx = tid; idx < count; idx += 256) {
      int tx = t0 + tt - padl;
      float v = 0.0f;
      if (ci < Cin && tx >= 0 && tx < Lin)
        v = xb[(size_t)ci * Lin + tx];
      sx[(size_t)tt * KP + ci] = (bf16_t)v;
      tt += 256;
      while (tt >= window) { tt -= window; ++ci; }
    }
  }
  __syncthreads();

  const int wv     = tid >> 5;
  const int lane   = tid & 31;
  const int hsel   = lane >> 4;   // K-half select (A: 8-chunks, B: 8-chunks)
  const int nm     = lane & 15;   // A: M row, B: N col
  const int coBase = blockIdx.y * 32 + (wv >> 2) * 16;
  int arow = coBase + nm; if (arow >= Cout) arow = Cout - 1;  // clamp (Cout<32)
  const int tL0 = (wv & 3) * 32;

  v8f acc0 = {0.f,0.f,0.f,0.f,0.f,0.f,0.f,0.f};
  v8f acc1 = acc0;

  for (int j = 0; j < K; ++j) {
    const bf16_t* wrow = w + ((size_t)j * Cout + arow) * (size_t)KP + hsel * 8;
    const int lt0 = tL0 + nm + j * dil;
    const bf16_t* xr0 = sx + (size_t)lt0 * KP + hsel * 8;
    const bf16_t* xr1 = xr0 + (size_t)16 * KP;
    #pragma unroll
    for (int kc = 0; kc < KP; kc += 32) {
      // A fragment (weights) straight from global, bf16
      v8bf a0 = *(const v8bf*)(wrow + kc);
      v8bf a1 = *(const v8bf*)(wrow + kc + 16);
      v16bf A = __builtin_shufflevector(a0, a1, 0,1,2,3,4,5,6,7,8,9,10,11,12,13,14,15);
      // B fragment 0 from LDS
      v8bf b0 = *(const v8bf*)(xr0 + kc);
      v8bf b1 = *(const v8bf*)(xr0 + kc + 16);
      v16bf B0 = __builtin_shufflevector(b0, b1, 0,1,2,3,4,5,6,7,8,9,10,11,12,13,14,15);
      acc0 = __builtin_amdgcn_wmma_f32_16x16x32_bf16(false, A, false, B0, (short)0, acc0, false, false);
      // B fragment 1 (A reuse)
      v8bf c0 = *(const v8bf*)(xr1 + kc);
      v8bf c1 = *(const v8bf*)(xr1 + kc + 16);
      v16bf B1 = __builtin_shufflevector(c0, c1, 0,1,2,3,4,5,6,7,8,9,10,11,12,13,14,15);
      acc1 = __builtin_amdgcn_wmma_f32_16x16x32_bf16(false, A, false, B1, (short)0, acc1, false, false);
    }
  }

  // Epilogue: bias + optional residual. C layout: VGPR r -> M = r + 8*hsel.
  #pragma unroll
  for (int fgi = 0; fgi < 2; ++fgi) {
    v8f acc = fgi ? acc1 : acc0;
    int tg = t0 + tL0 + fgi * 16 + nm;
    if (tg < Lq) {
      size_t obase = ((size_t)b * Cout) * (size_t)LoutFull + (size_t)tg * ostride + ooff;
      #pragma unroll
      for (int r = 0; r < 8; ++r) {
        int co = coBase + r + 8 * hsel;
        if (co < Cout) {
          float v = acc[r];
          if (bias) v += bias[co];
          size_t oi = obase + (size_t)co * LoutFull;
          if (res) v += res[oi];
          out[oi] = v;
        }
      }
    }
  }
}

// ---------------- fused 2x FIR upsample + SnakeBeta ------------------------
// grid: (ceil(L/256), C, B) -- no div/mod per element.
// y[2s]   = snake( 2*sum_i F[11-2i]*x[clamp(s+i-3)] )
// y[2s+1] = snake( 2*sum_i F[10-2i]*x[clamp(s+i-2)] )
__global__ __launch_bounds__(256) void up2snake_k(
    const float* __restrict__ x, const float* __restrict__ la,
    const float* __restrict__ lb, float* __restrict__ y, int L, Filt F)
{
  int s = blockIdx.x * 256 + threadIdx.x;
  if (s >= L) return;
  int c = blockIdx.y;
  int b = blockIdx.z;
  int C = gridDim.y;
  const float* row = x + ((size_t)b * C + c) * L;
  float xv[7];
  #pragma unroll
  for (int i = 0; i < 7; ++i) {
    int t = s - 3 + i;
    t = t < 0 ? 0 : (t >= L ? L - 1 : t);
    xv[i] = row[t];
  }
  float ev = 2.0f * (F.f[11]*xv[0] + F.f[9]*xv[1] + F.f[7]*xv[2] +
                     F.f[5]*xv[3]  + F.f[3]*xv[4] + F.f[1]*xv[5]);
  float od = 2.0f * (F.f[10]*xv[1] + F.f[8]*xv[2] + F.f[6]*xv[3] +
                     F.f[4]*xv[4]  + F.f[2]*xv[5] + F.f[0]*xv[6]);
  float a  = __expf(la[c]);
  float bi = 1.0f / (__expf(lb[c]) + 1e-9f);
  float se = __sinf(ev * a), so = __sinf(od * a);
  float* yr = y + ((size_t)b * C + c) * (size_t)(2 * L);
  yr[2*s]   = ev + se * se * bi;
  yr[2*s+1] = od + so * so * bi;
}

// ---------------- 2x FIR downsample (stride 2, edge pad (5,6)) -------------
// grid: (ceil(L/256), C, B); L = out length, input length 2L.
__global__ __launch_bounds__(256) void down2_k(
    const float* __restrict__ x, float* __restrict__ y, int L, Filt F)
{
  int s = blockIdx.x * 256 + threadIdx.x;
  if (s >= L) return;
  int c = blockIdx.y;
  int b = blockIdx.z;
  int C = gridDim.y;
  int L2 = 2 * L;
  const float* row = x + ((size_t)b * C + c) * (size_t)L2;
  float acc = 0.0f;
  #pragma unroll
  for (int j = 0; j < 12; ++j) {
    int t = 2 * s + j - 5;
    t = t < 0 ? 0 : (t >= L2 ? L2 - 1 : t);
    acc += F.f[j] * row[t];
  }
  y[((size_t)b * C + c) * L + s] = acc;
}

// out = alpha * (in0 + in1?)   (copy / accumulate / scale)
__global__ __launch_bounds__(256) void axpy_k(
    const float* __restrict__ in0, const float* __restrict__ in1,
    float* __restrict__ out, float alpha, int n)
{
  int idx = blockIdx.x * 256 + threadIdx.x;
  if (idx >= n) return;
  float v = in0[idx];
  if (in1) v += in1[idx];
  out[idx] = alpha * v;
}

__global__ __launch_bounds__(256) void tanh_k(
    const float* __restrict__ in, float* __restrict__ out, int n)
{
  int idx = blockIdx.x * 256 + threadIdx.x;
  if (idx < n) out[idx] = tanhf(in[idx]);
}

// ---------------------------------------------------------------------------
static double bessel_i0(double x) {
  double s = 1.0, t = 1.0;
  for (int k = 1; k <= 40; ++k) { double u = x / (2.0 * k); t *= u * u; s += t; }
  return s;
}

extern "C" void kernel_launch(void* const* d_in, const int* in_sizes, int n_in,
                              void* d_out, int out_size, void* d_ws, size_t ws_size,
                              hipStream_t stream)
{
  (void)in_sizes; (void)n_in; (void)out_size; (void)ws_size;
  const double PI = 3.14159265358979323846;

  // ---- kaiser-sinc 12-tap filter (matches reference _fir12) ----
  Filt F;
  {
    const int Kf = 12, half = 6;
    double A = 2.285 * (half - 1) * PI * (4.0 * 0.3) + 7.95;
    double beta = (A > 50.0) ? 0.1102 * (A - 8.7)
                : (A >= 21.0 ? 0.5842 * pow(A - 21.0, 0.4) + 0.07886 * (A - 21.0) : 0.0);
    double i0b = bessel_i0(beta);
    double f[12], sum = 0.0;
    for (int nn = 0; nn < Kf; ++nn) {
      double ratio = (2.0 * nn) / (Kf - 1) - 1.0;
      double wk = bessel_i0(beta * sqrt(fmax(0.0, 1.0 - ratio * ratio))) / i0b;
      double t  = (nn - half) + 0.5;
      double xx = 0.5 * t;                       // 2*cutoff*t, cutoff = 0.25
      double sc = (xx == 0.0) ? 1.0 : sin(PI * xx) / (PI * xx);
      f[nn] = 0.5 * wk * sc;
      sum += f[nn];
    }
    for (int nn = 0; nn < Kf; ++nn) F.f[nn] = (float)(f[nn] / sum);
  }

  // ---- input leaf map (JAX pytree order: sorted dict keys; params then x) --
  auto P32 = [&](int i) -> const float* { return (const float*)d_in[i]; };
  const int IPOST_A = 0, IPOST_B = 1, IPOST_W = 2, IPRE_B = 3, IPRE_W = 4;
  auto resBase = [&](int s, int j) { return 5 + (s * 3 + j) * 24; };
  const int IUPS = 293, IX = 301;

  // ---- workspace bump allocator ----
  size_t off = 0;
  auto balloc = [&](size_t bytes) -> char* {
    size_t a = (off + 255) & ~(size_t)255;
    off = a + bytes;
    return (char*)d_ws + a;
  };
  const size_t ACT_ELEMS = (size_t)4 * 4194304;          // max B*C*L = 16.8M
  const size_t ACT = ACT_ELEMS * sizeof(float);          // 64 MiB
  float* RX  = (float*)balloc(ACT);
  float* RXS = (float*)balloc(ACT);
  float* RHA = (float*)balloc(ACT);
  float* RHB = (float*)balloc(ACT);
  float* RT1 = (float*)balloc(ACT);
  float* RUP = (float*)balloc(2 * ACT);

  auto rup32 = [](int v) { return (v + 31) & ~31; };

  auto repack_conv = [&](const float* src, int Co, int Ci, int k) -> bf16_t* {
    int KP = rup32(Ci);
    int n = k * Co * KP;
    bf16_t* dst = (bf16_t*)balloc((size_t)n * sizeof(bf16_t));
    repack_conv_k<<<dim3((n + 255) / 256), dim3(256), 0, stream>>>(src, dst, Co, Ci, KP, k, n);
    return dst;
  };
  auto repack_tconv = [&](const float* wt, int Ci, int Co, int k, int u, int pad) -> bf16_t* {
    int KP = rup32(Ci); int P = k / u;
    int n = u * P * Co * KP;
    bf16_t* dst = (bf16_t*)balloc((size_t)n * sizeof(bf16_t));
    repack_tconv_k<<<dim3((n + 255) / 256), dim3(256), 0, stream>>>(wt, dst, Ci, Co, KP, k, u, P, pad, n);
    return dst;
  };

  const int B = 4;
  auto conv = [&](const float* x, const bf16_t* w, const float* bias, const float* res,
                  float* out, int Cin, int Lin, int Cout, int K, int dil, int padl,
                  int Lq, int ostride, int ooff, int LoutFull) {
    int KP = rup32(Cin);
    int window = 128 + (K - 1) * dil;
    size_t smem = (size_t)window * KP * sizeof(bf16_t);
    dim3 g((Lq + 127) / 128, (Cout + 31) / 32, B);
    switch (KP) {
      case 32:
        conv_wmma_k<32><<<g, dim3(256), smem, stream>>>(x, w, bias, res, out, Cin, Lin,
                                                        Cout, K, dil, padl, Lq, ostride, ooff, LoutFull);
        break;
      case 64:
        conv_wmma_k<64><<<g, dim3(256), smem, stream>>>(x, w, bias, res, out, Cin, Lin,
                                                        Cout, K, dil, padl, Lq, ostride, ooff, LoutFull);
        break;
      case 128:
        conv_wmma_k<128><<<g, dim3(256), smem, stream>>>(x, w, bias, res, out, Cin, Lin,
                                                         Cout, K, dil, padl, Lq, ostride, ooff, LoutFull);
        break;
      case 256:
        conv_wmma_k<256><<<g, dim3(256), smem, stream>>>(x, w, bias, res, out, Cin, Lin,
                                                         Cout, K, dil, padl, Lq, ostride, ooff, LoutFull);
        break;
      default:
        conv_wmma_k<512><<<g, dim3(256), smem, stream>>>(x, w, bias, res, out, Cin, Lin,
                                                         Cout, K, dil, padl, Lq, ostride, ooff, LoutFull);
        break;
    }
  };
  auto up2snake = [&](const float* x, const float* la, const float* lb, float* y, int C, int L) {
    dim3 g((L + 255) / 256, C, B);
    up2snake_k<<<g, dim3(256), 0, stream>>>(x, la, lb, y, L, F);
  };
  auto down2 = [&](const float* x, float* y, int C, int L) {
    dim3 g((L + 255) / 256, C, B);
    down2_k<<<g, dim3(256), 0, stream>>>(x, y, L, F);
  };
  auto axpy = [&](const float* a, const float* bb, float* o, float alpha, int n) {
    axpy_k<<<dim3((n + 255) / 256), dim3(256), 0, stream>>>(a, bb, o, alpha, n);
  };

  // ---- model shapes ----
  int chin_a[4] = {512, 256, 128, 64};
  int ch_a[4]   = {256, 128, 64, 32};
  int U_a[4]    = {8, 8, 2, 2};
  int K_a[4]    = {16, 16, 4, 4};
  int RKa[3]    = {3, 7, 11};
  int RDa[3]    = {1, 3, 5};

  // ---- repack all weights to bf16 (every call; deterministic) ----
  bf16_t* Wpre = repack_conv(P32(IPRE_W), 512, 100, 7);
  bf16_t* Wups[4]; bf16_t* Wc1[4][3][3]; bf16_t* Wc2[4][3][3];
  for (int s = 0; s < 4; ++s) {
    int pad = K_a[s] - 1 - (K_a[s] - U_a[s]) / 2;
    Wups[s] = repack_tconv(P32(IUPS + 2 * s), chin_a[s], ch_a[s], K_a[s], U_a[s], pad);
    for (int j = 0; j < 3; ++j) {
      int rb = resBase(s, j); int k = RKa[j]; int c = ch_a[s];
      for (int i = 0; i < 3; ++i) {
        Wc1[s][j][i] = repack_conv(P32(rb + 12 + 2 * i), c, c, k);
        Wc2[s][j][i] = repack_conv(P32(rb + 18 + 2 * i), c, c, k);
      }
    }
  }
  bf16_t* Wpost = repack_conv(P32(IPOST_W), 1, 32, 7);

  // ---- forward ----
  // pre conv: x(4,100,512) -> RX(4,512,512), pad 3
  conv((const float*)d_in[IX], Wpre, P32(IPRE_B), nullptr, RX,
       100, 512, 512, 7, 1, 3, 512, 1, 0, 512);

  int L = 512;
  for (int s = 0; s < 4; ++s) {
    int Ci = chin_a[s], Co = ch_a[s], u = U_a[s], kk = K_a[s];
    int pad = kk - 1 - (kk - u) / 2;
    int P = kk / u;
    int Lout = L * u;
    // transposed conv as u phase-convs (2 taps each), output stride u
    for (int r = 0; r < u; ++r) {
      int j0 = ((pad - r) % u + u) % u;
      int e0 = (r + j0 - pad) / u;              // exact: 0 or -1
      bf16_t* wph = Wups[s] + (size_t)r * P * Co * rup32(Ci);
      conv(RX, wph, P32(IUPS + 2 * s + 1), nullptr, RT1,
           Ci, L, Co, P, 1, -e0, L, u, r, Lout);
    }
    { float* t = RX; RX = RT1; RT1 = t; }
    L = Lout;
    int n = B * Co * L;

    // three AMP residual blocks, averaged
    for (int j = 0; j < 3; ++j) {
      int rb = resBase(s, j); int k = RKa[j];
      const float* h = RX;
      float* scr[2] = {RHA, RHB};
      int sel = 0;
      for (int i = 0; i < 3; ++i) {
        int d = RDa[i];
        float* fr = scr[sel];
        up2snake(h, P32(rb + 4 * i), P32(rb + 4 * i + 1), RUP, Co, L);
        down2(RUP, RT1, Co, L);
        conv(RT1, Wc1[s][j][i], P32(rb + 13 + 2 * i), nullptr, fr,
             Co, L, Co, k, d, (k - 1) * d / 2, L, 1, 0, L);
        up2snake(fr, P32(rb + 4 * i + 2), P32(rb + 4 * i + 3), RUP, Co, L);
        down2(RUP, RT1, Co, L);
        conv(RT1, Wc2[s][j][i], P32(rb + 19 + 2 * i), h, fr,
             Co, L, Co, k, 1, (k - 1) / 2, L, 1, 0, L);
        h = fr; sel ^= 1;
      }
      if (j == 0)      axpy(h, nullptr, RXS, 1.0f, n);
      else if (j == 1) axpy(RXS, h, RXS, 1.0f, n);
      else             axpy(RXS, h, RX, 1.0f / 3.0f, n);
    }
  }

  // post: act1d -> conv_post (Cout=1, no bias) -> tanh
  up2snake(RX, P32(IPOST_A), P32(IPOST_B), RUP, 32, L);
  down2(RUP, RT1, 32, L);
  conv(RT1, Wpost, nullptr, nullptr, RHA, 32, L, 1, 7, 1, 3, L, 1, 0, L);
  int nOut = B * L;  // 524288
  tanh_k<<<dim3((nOut + 255) / 256), dim3(256), 0, stream>>>(RHA, (float*)d_out, nOut);
}
